// LinearMem_24008867185110
// MI455X (gfx1250) — matrix operand order repaired and checked by reading, not compiled, not run on hardware
//
#include <hip/hip_runtime.h>

// ---------------------------------------------------------------------------
// Blockwise-int8 quant-dequant + GEMM + bias for MI455X (gfx1250, wave32).
//   reference: xq(1x32 blocks) @ wq(32x32 blocks, of weight^T) + bias
// Strategy: dequantize both operands once into f16 (quant levels are exact in
// f16), pre-packed into WMMA fragment layout; GEMM via v_wmma_f32_16x16x32_f16
// with f32 accumulators. Packed operands (100 MB) are L2-resident (192 MB L2).
// Wave tile 64Mx64N balances fragment bandwidth (128 B/cycle/SIMD) against the
// ~4-cycle f16 WMMA issue rate.  WGP-scope global_prefetch_b8, all offsets as
// 24-bit immediates off the live loop pointers (no address VALU, no hazards),
// covers all 8 fragment panels 8..11 chunks ahead.
// ---------------------------------------------------------------------------

typedef __attribute__((ext_vector_type(16))) _Float16 v16h;
typedef __attribute__((ext_vector_type(8)))  float    v8f;

constexpr int M  = 8192;
constexpr int N  = 4096;
constexpr int K  = 4096;
constexpr int KC = K / 32;   // 128 k-chunks of 32
constexpr int NB = N / 32;   // 128 n-blocks for weight scales

// Workspace layout (bytes)
constexpr size_t QA_BYTES = (size_t)M * K * 2;  // packed A fragments (f16)
constexpr size_t QB_BYTES = (size_t)N * K * 2;  // packed B fragments (f16)
constexpr size_t QA_OFF   = 0;
constexpr size_t QB_OFF   = QA_BYTES;
constexpr size_t SW_OFF   = QA_BYTES + QB_BYTES;        // KC*NB floats

// WGP-scope prefetch (default scope), displacement in the 24-bit imm offset:
// consumes the live pointer VGPR pair directly.  Per-lane addresses (base
// includes lane*32) make one op cover one 1KB fragment panel chunk.
#define PF_WGP(ptr, off_lit)                                            \
  asm volatile("global_prefetch_b8 %0, off offset:" off_lit             \
               :: "v"((unsigned long long)(ptr)))

// All 4 panels of one operand (stride TS=131072), chunks +8..+11 ahead (+8192..+11264).
#define PF_PANELS(p)                                                    \
  do {                                                                  \
    PF_WGP(p, "8192");   PF_WGP(p, "9216");                             \
    PF_WGP(p, "10240");  PF_WGP(p, "11264");                            \
    PF_WGP(p, "139264"); PF_WGP(p, "140288");                           \
    PF_WGP(p, "141312"); PF_WGP(p, "142336");                           \
    PF_WGP(p, "270336"); PF_WGP(p, "271360");                           \
    PF_WGP(p, "272384"); PF_WGP(p, "273408");                           \
    PF_WGP(p, "401408"); PF_WGP(p, "402432");                           \
    PF_WGP(p, "403456"); PF_WGP(p, "404480");                           \
  } while (0)

// ---------------------------------------------------------------------------
// Kernel 1: per-(32k x 32n) block max|w| of weight^T.  One wave per block.
// ---------------------------------------------------------------------------
__global__ __launch_bounds__(256) void scale_w_kernel(
    const float* __restrict__ w, float* __restrict__ swmax) {
  const int lane = threadIdx.x & 31;
  const int idx  = blockIdx.x * 8 + (threadIdx.x >> 5);  // over KC*NB blocks
  const int nb   = idx & (NB - 1);
  const int kb   = idx >> 7;                             // idx / NB
  const float* src = w + (size_t)(nb * 32 + lane) * K + kb * 32;
  float mx = 0.0f;
#pragma unroll
  for (int i = 0; i < 32; i += 4) {
    float4 t = *(const float4*)(src + i);
    mx = fmaxf(mx, fmaxf(fmaxf(fabsf(t.x), fabsf(t.y)),
                         fmaxf(fabsf(t.z), fabsf(t.w))));
  }
#pragma unroll
  for (int off = 16; off > 0; off >>= 1)
    mx = fmaxf(mx, __shfl_xor(mx, off, 32));
  if (lane == 0) swmax[idx] = fmaxf(mx, 1e-12f);
}

// ---------------------------------------------------------------------------
// Kernel 2: quant-dequant weight^T to f16 and pack into WMMA B-fragment tiles.
// Tile = 32(K) x 16(N) f16, 1024 B, indexed [ntile][kc]; within a tile,
// lane n%16 (K 0..7,16..23) and lane n%16+16 (K 8..15,24..31) each hold 32 contiguous
// bytes at lane*32.
// ---------------------------------------------------------------------------
__global__ __launch_bounds__(256) void pack_w_kernel(
    const float* __restrict__ w, const float* __restrict__ swmax,
    _Float16* __restrict__ qB) {
  const int id = blockIdx.x * 256 + threadIdx.x;  // over N*KC, kb fastest
  const int kb = id & (KC - 1);
  const int n  = id >> 7;
  const float* src = w + (size_t)n * K + kb * 32;
  const float mx = swmax[kb * NB + (n >> 5)];
  const float qs = 127.0f / mx;
  const float dq = mx * (1.0f / 127.0f);
  _Float16 d[32];
#pragma unroll
  for (int i = 0; i < 32; i += 4) {
    float4 t = *(const float4*)(src + i);
    d[i + 0] = (_Float16)(rintf(t.x * qs) * dq);
    d[i + 1] = (_Float16)(rintf(t.y * qs) * dq);
    d[i + 2] = (_Float16)(rintf(t.z * qs) * dq);
    d[i + 3] = (_Float16)(rintf(t.w * qs) * dq);
  }
  _Float16* tile = qB + (((size_t)(n >> 4) * KC + kb) << 9);  // *512 halves
  union H16 { _Float16 h[16]; float4 f[2]; } lo, hi;
#pragma unroll
  for (int j = 0; j < 16; ++j) { lo.h[j] = d[(j & 7) + ((j >> 3) << 4)]; hi.h[j] = d[8 + (j & 7) + ((j >> 3) << 4)]; }
  float4* t0 = (float4*)(tile + (size_t)(n & 15) * 16);
  float4* t1 = (float4*)(tile + (size_t)((n & 15) + 16) * 16);
  t0[0] = lo.f[0]; t0[1] = lo.f[1];
  t1[0] = hi.f[0]; t1[1] = hi.f[1];
}

// ---------------------------------------------------------------------------
// Kernel 3: quant-dequant x (per-row 1x32 blocks: scale computed inline from
// exactly this thread's 32 values) and pack into WMMA A-fragment tiles.
// 16-bit A 16x32 layout (ISA 7.12.2): lane m%16 (hi=0) holds K {0..7,16..23},
// lane m%16+16 (hi=1) holds K {8..15,24..31}, 32 contiguous bytes per lane.
// ---------------------------------------------------------------------------
__global__ __launch_bounds__(256) void pack_x_kernel(
    const float* __restrict__ x, _Float16* __restrict__ qA) {
  const int id = blockIdx.x * 256 + threadIdx.x;  // over M*KC, kb fastest
  const int kb = id & (KC - 1);
  const int m  = id >> 7;
  const float* src = x + (size_t)m * K + kb * 32;
  float v[32];
  float mx = 0.0f;
#pragma unroll
  for (int i = 0; i < 32; i += 4) {
    float4 t = *(const float4*)(src + i);
    v[i] = t.x; v[i + 1] = t.y; v[i + 2] = t.z; v[i + 3] = t.w;
    mx = fmaxf(mx, fmaxf(fmaxf(fabsf(t.x), fabsf(t.y)),
                         fmaxf(fabsf(t.z), fabsf(t.w))));
  }
  mx = fmaxf(mx, 1e-12f);
  const float qs = 127.0f / mx;
  const float dq = mx * (1.0f / 127.0f);
  _Float16 d[32];
#pragma unroll
  for (int i = 0; i < 32; ++i) d[i] = (_Float16)(rintf(v[i] * qs) * dq);
  _Float16* tile = qA + (((size_t)(m >> 4) * KC + kb) << 9);
  union H16 { _Float16 h[16]; float4 f[2]; } lo, hi;
#pragma unroll
  for (int j = 0; j < 8; ++j) {
    lo.h[j] = d[j];      lo.h[8 + j] = d[16 + j];  // hi=0: K 0..7,16..23
    hi.h[j] = d[8 + j];  hi.h[8 + j] = d[24 + j];  // hi=1: K 8..15,24..31
  }
  float4* t0 = (float4*)(tile + (size_t)(m & 15) * 16);
  float4* t1 = (float4*)(tile + (size_t)((m & 15) + 16) * 16);
  t0[0] = lo.f[0]; t0[1] = lo.f[1];
  t1[0] = hi.f[0]; t1[1] = hi.f[1];
}

// ---------------------------------------------------------------------------
// Kernel 4: GEMM.  Wave tile 64M x 64N (4x4 WMMA subtiles, 128 f32 acc VGPRs),
// workgroup = 8 waves (2M x 4N) -> 128M x 256N.  Manual 4-chunk unroll:
// per iteration, 32 WGP-scope prefetches (imm offsets off the live pointers),
// then 4 x { 16 b128 fragment loads + 16 v_wmma_f32_16x16x32_f16 }.
// Fragment traffic = 8 KB / 16 WMMAs ~= 128 B/cycle/SIMD; operands stream
// from the 192 MB L2 via WGP$/L0 prefetched 8..11 chunks ahead.
// ---------------------------------------------------------------------------
__global__ __launch_bounds__(256) void gemm_kernel(
    const _Float16* __restrict__ qA, const _Float16* __restrict__ qB,
    const float* __restrict__ bias, float* __restrict__ out) {
  const int lane = threadIdx.x & 31;
  const int wave = threadIdx.x >> 5;
  const int wm = wave & 1;
  const int wn = wave >> 1;
  const int m0 = blockIdx.x * 128 + wm * 64;
  const int n0 = blockIdx.y * 256 + wn * 64;
  constexpr size_t TS = (size_t)KC * 1024;  // 131072 bytes per fragment panel

  const char* pA = (const char*)qA + (size_t)(m0 >> 4) * TS + (size_t)lane * 32;
  const char* pB = (const char*)qB + (size_t)(n0 >> 4) * TS + (size_t)lane * 32;

  v8f acc[4][4] = {};

  for (int kc = 0; kc < KC; kc += 4) {
    // Prefetch all 8 panels for the 4 chunks 8..11 ahead; every offset is a
    // compile-time immediate off pA/pB -> zero address arithmetic.
    PF_PANELS(pA);
    PF_PANELS(pB);

#pragma unroll
    for (int c = 0; c < 4; ++c) {
      const char* qa = pA + (size_t)c * 1024;
      const char* qb = pB + (size_t)c * 1024;
      v16h a0 = *(const v16h*)(qa);
      v16h a1 = *(const v16h*)(qa + TS);
      v16h a2 = *(const v16h*)(qa + 2 * TS);
      v16h a3 = *(const v16h*)(qa + 3 * TS);
      v16h b0 = *(const v16h*)(qb);
      v16h b1 = *(const v16h*)(qb + TS);
      v16h b2 = *(const v16h*)(qb + 2 * TS);
      v16h b3 = *(const v16h*)(qb + 3 * TS);

      acc[0][0] = __builtin_amdgcn_wmma_f32_16x16x32_f16(false, a0, false, b0,
                                                         (short)0, acc[0][0], false, false);
      acc[0][1] = __builtin_amdgcn_wmma_f32_16x16x32_f16(false, a0, false, b1,
                                                         (short)0, acc[0][1], false, false);
      acc[0][2] = __builtin_amdgcn_wmma_f32_16x16x32_f16(false, a0, false, b2,
                                                         (short)0, acc[0][2], false, false);
      acc[0][3] = __builtin_amdgcn_wmma_f32_16x16x32_f16(false, a0, false, b3,
                                                         (short)0, acc[0][3], false, false);
      acc[1][0] = __builtin_amdgcn_wmma_f32_16x16x32_f16(false, a1, false, b0,
                                                         (short)0, acc[1][0], false, false);
      acc[1][1] = __builtin_amdgcn_wmma_f32_16x16x32_f16(false, a1, false, b1,
                                                         (short)0, acc[1][1], false, false);
      acc[1][2] = __builtin_amdgcn_wmma_f32_16x16x32_f16(false, a1, false, b2,
                                                         (short)0, acc[1][2], false, false);
      acc[1][3] = __builtin_amdgcn_wmma_f32_16x16x32_f16(false, a1, false, b3,
                                                         (short)0, acc[1][3], false, false);
      acc[2][0] = __builtin_amdgcn_wmma_f32_16x16x32_f16(false, a2, false, b0,
                                                         (short)0, acc[2][0], false, false);
      acc[2][1] = __builtin_amdgcn_wmma_f32_16x16x32_f16(false, a2, false, b1,
                                                         (short)0, acc[2][1], false, false);
      acc[2][2] = __builtin_amdgcn_wmma_f32_16x16x32_f16(false, a2, false, b2,
                                                         (short)0, acc[2][2], false, false);
      acc[2][3] = __builtin_amdgcn_wmma_f32_16x16x32_f16(false, a2, false, b3,
                                                         (short)0, acc[2][3], false, false);
      acc[3][0] = __builtin_amdgcn_wmma_f32_16x16x32_f16(false, a3, false, b0,
                                                         (short)0, acc[3][0], false, false);
      acc[3][1] = __builtin_amdgcn_wmma_f32_16x16x32_f16(false, a3, false, b1,
                                                         (short)0, acc[3][1], false, false);
      acc[3][2] = __builtin_amdgcn_wmma_f32_16x16x32_f16(false, a3, false, b2,
                                                         (short)0, acc[3][2], false, false);
      acc[3][3] = __builtin_amdgcn_wmma_f32_16x16x32_f16(false, a3, false, b3,
                                                         (short)0, acc[3][3], false, false);
    }
    pA += 4096;
    pB += 4096;
  }

  // Epilogue: C/D layout -> lanes 0-15: M = r, N = lane; lanes 16-31: M = r+8.
  const int hi = lane >> 4;
  const int cl = lane & 15;
#pragma unroll
  for (int mf = 0; mf < 4; ++mf) {
#pragma unroll
    for (int nf = 0; nf < 4; ++nf) {
      const int col = n0 + nf * 16 + cl;
      const float bv = bias[col];
      float* op = out + (size_t)(m0 + mf * 16 + hi * 8) * N + col;
#pragma unroll
      for (int r = 0; r < 8; ++r) op[(size_t)r * N] = acc[mf][nf][r] + bv;
    }
  }
}

// ---------------------------------------------------------------------------
extern "C" void kernel_launch(void* const* d_in, const int* in_sizes, int n_in,
                              void* d_out, int out_size, void* d_ws, size_t ws_size,
                              hipStream_t stream) {
  const float* x    = (const float*)d_in[0];  // (M, K) f32
  const float* w    = (const float*)d_in[1];  // (N, K) f32
  const float* bias = (const float*)d_in[2];  // (N,)  f32
  float* out        = (float*)d_out;          // (M, N) f32

  _Float16* qA   = (_Float16*)((char*)d_ws + QA_OFF);
  _Float16* qB   = (_Float16*)((char*)d_ws + QB_OFF);
  float*    swmx = (float*)((char*)d_ws + SW_OFF);

  // 1) weight^T 32x32 block scales: KC*NB blocks, one wave each, 8 waves/WG.
  scale_w_kernel<<<dim3((KC * NB) / 8), dim3(256), 0, stream>>>(w, swmx);
  // 2) pack dequantized weight^T into B fragments: N*KC threads.
  pack_w_kernel<<<dim3((N * KC) / 256), dim3(256), 0, stream>>>(w, swmx, qB);
  // 3) pack dequantized x into A fragments (scale fused): M*KC threads.
  pack_x_kernel<<<dim3((M * KC) / 256), dim3(256), 0, stream>>>(x, qA);
  // 4) WMMA GEMM + bias: grid (M/128, N/256), 256 threads (8 waves, 2Mx4N).
  gemm_kernel<<<dim3(M / 128, N / 256), dim3(256), 0, stream>>>(qA, qB, bias, out);
}
